// AdaptiveLayoutEncoder_1391569404370
// MI455X (gfx1250) — matrix operand order repaired and checked by reading, not compiled, not run on hardware
//
#include <hip/hip_runtime.h>
#include <hip/hip_bf16.h>
#include <cstdint>

typedef _Float16 f16;
typedef _Float16 h16v __attribute__((ext_vector_type(16)));
typedef float    f8v  __attribute__((ext_vector_type(8)));

#define HIDN  128
#define HEADS 4
#define HC    32

// ---------- order-preserving float <-> u32 mapping for atomic max ----------
__device__ __forceinline__ unsigned fenc(float f) {
  unsigned u = __float_as_uint(f);
  return (u & 0x80000000u) ? ~u : (u | 0x80000000u);
}
__device__ __forceinline__ float fdec(unsigned u) {
  return __uint_as_float((u & 0x80000000u) ? (u ^ 0x80000000u) : ~u);
}

// ---------- fused relu + LayerNorm over a 128-wide row (blockDim=128) ------
__device__ __forceinline__ f16 relu_ln_val(float x, const float* __restrict__ g,
                                           const float* __restrict__ b, int t) {
  x = fmaxf(x, 0.f);
  __shared__ float red[4];
  float s = x;
  #pragma unroll
  for (int o = 16; o > 0; o >>= 1) s += __shfl_down(s, o, 32);
  if ((t & 31) == 0) red[t >> 5] = s;
  __syncthreads();
  const float mean = (red[0] + red[1] + red[2] + red[3]) * (1.f / 128.f);
  __syncthreads();
  const float d = x - mean;
  float s2 = d * d;
  #pragma unroll
  for (int o = 16; o > 0; o >>= 1) s2 += __shfl_down(s2, o, 32);
  if ((t & 31) == 0) red[t >> 5] = s2;
  __syncthreads();
  const float var = (red[0] + red[1] + red[2] + red[3]) * (1.f / 128.f);
  return (f16)(d * rsqrtf(var + 1e-5f) * g[t] + b[t]);
}

// ---------- WMMA GEMM:  C[M x 128] = A[M x K](f16,row-major) * Wt[128 x K]^T + bias
// Wt is stored transposed: Wt[n][k]. Block = 256 thr (8 waves), tile 32 x 128.
__global__ __launch_bounds__(256)
void wmma_gemm128(const f16* __restrict__ A, int lda,
                  const f16* __restrict__ Bt, int ldb,
                  const float* __restrict__ bias,
                  float* __restrict__ Cf, f16* __restrict__ Ch, int ldc,
                  int K, int M) {
  const int lane = threadIdx.x & 31;
  const int wave = threadIdx.x >> 5;
  const int r0 = blockIdx.x * 32 + (wave & 1) * 16;
  const int c0 = (wave >> 1) * 32;
  const int hi = lane >> 4;   // 0/1
  const int lo = lane & 15;
  const int arow = min(r0 + lo, M - 1);
  const f16* ap  = A  + (size_t)arow * lda + hi * 16;
  const f16* bp0 = Bt + (size_t)(c0 + lo) * ldb + hi * 16;
  const f16* bp1 = bp0 + (size_t)16 * ldb;
  f8v acc0 = {}; f8v acc1 = {};
  for (int k0 = 0; k0 < K; k0 += 32) {
    h16v a, b0, b1;
    ((uint4*)&a)[0]  = *(const uint4*)(ap + k0);        // K = kb..kb+7
    ((uint4*)&a)[1]  = *(const uint4*)(ap + k0 + 8);    // K = kb+8..kb+15
    ((uint4*)&b0)[0] = *(const uint4*)(bp0 + k0);
    ((uint4*)&b0)[1] = *(const uint4*)(bp0 + k0 + 8);
    ((uint4*)&b1)[0] = *(const uint4*)(bp1 + k0);
    ((uint4*)&b1)[1] = *(const uint4*)(bp1 + k0 + 8);
    acc0 = __builtin_amdgcn_wmma_f32_16x16x32_f16(false, a, false, b0, (short)0, acc0, false, false);
    acc1 = __builtin_amdgcn_wmma_f32_16x16x32_f16(false, a, false, b1, (short)0, acc1, false, false);
  }
  const float bb0 = bias ? bias[c0 + lo]      : 0.f;
  const float bb1 = bias ? bias[c0 + 16 + lo] : 0.f;
  #pragma unroll
  for (int r = 0; r < 8; ++r) {
    const int row = r0 + r + hi * 8;          // C layout: M = r + 8*(lane/16)
    if (row >= M) continue;
    const size_t o0 = (size_t)row * ldc + c0 + lo;
    const float v0 = acc0[r] + bb0;
    const float v1 = acc1[r] + bb1;
    if (Cf) { Cf[o0] = v0; Cf[o0 + 16] = v1; }
    if (Ch) { Ch[o0] = (f16)v0; Ch[o0 + 16] = (f16)v1; }
  }
}

// ---------- weight prep: transpose f32 W[K0 x 128] -> f16 Wt[128 x Kp], zero-pad K
__global__ __launch_bounds__(HIDN)
void prep_wt(const float* __restrict__ W, int K0, int Kp, f16* __restrict__ Wt) {
  const int n = blockIdx.x;                 // output channel 0..127
  for (int k = threadIdx.x; k < Kp; k += blockDim.x)
    Wt[(size_t)n * Kp + k] = (f16)(k < K0 ? W[(size_t)k * HIDN + n] : 0.f);
}

// ---------- flow matrix -> f16, K padded 100 -> 128 ----------
__global__ __launch_bounds__(HIDN)
void pad_flow_f16(const float* __restrict__ flow, f16* __restrict__ out, int N) {
  const int row = blockIdx.x, t = threadIdx.x;
  out[(size_t)row * HIDN + t] = (f16)(t < N ? flow[(size_t)row * N + t] : 0.f);
}

// ---------- dept encoder: d = LN(relu(dept @ dw + db)), written into cat[:, :128]
__global__ __launch_bounds__(HIDN)
void dept_encode(const float* __restrict__ dept, const float* __restrict__ dw,
                 const float* __restrict__ db, const float* __restrict__ dg,
                 const float* __restrict__ dbb, f16* __restrict__ Y, int ldy, int ATTR) {
  const int row = blockIdx.x, t = threadIdx.x;
  const float* a = dept + (size_t)row * ATTR;
  float x = db[t];
  for (int i = 0; i < ATTR; ++i) x += a[i] * dw[i * HIDN + t];
  Y[(size_t)row * ldy + t] = relu_ln_val(x, dg, dbb, t);
}

// ---------- generic relu+LN over f32 rows -> f16 ----------
__global__ __launch_bounds__(HIDN)
void relu_ln128(const float* __restrict__ X, const float* __restrict__ g,
                const float* __restrict__ b, f16* __restrict__ Y, int ldy) {
  const int row = blockIdx.x, t = threadIdx.x;
  Y[(size_t)row * ldy + t] = relu_ln_val(X[(size_t)row * HIDN + t], g, b, t);
}

// ---------- multi-head attention over one query row (K=32 dots: VALU) ------
__global__ __launch_bounds__(HIDN)
void attn128(const f16* __restrict__ Q, const f16* __restrict__ Kc,
             const f16* __restrict__ V, const int* __restrict__ mask,
             f16* __restrict__ Att, int N) {
  const int row = blockIdx.x;              // b*N + qi
  const int b = row / N;
  const int t = threadIdx.x;
  __shared__ float qs[HIDN];
  __shared__ float sc[HEADS][128];
  qs[t] = (float)Q[(size_t)row * HIDN + t];
  __syncthreads();
  const float scale = 0.17677669529663687f;   // 1/sqrt(32)
  for (int j = t; j < N; j += HIDN) {
    const f16* kr = Kc + ((size_t)b * N + j) * HIDN;
    const bool pad = (mask[b * N + j] == 0);
    #pragma unroll
    for (int h = 0; h < HEADS; ++h) {
      float dot = 0.f;
      #pragma unroll
      for (int c = 0; c < HC; ++c) dot += qs[h * HC + c] * (float)kr[h * HC + c];
      sc[h][j] = pad ? -1e9f : dot * scale;
    }
  }
  __syncthreads();
  const int h = t >> 5, ln = t & 31;        // wave h owns head h
  float m = -1e30f;
  for (int j = ln; j < N; j += 32) m = fmaxf(m, sc[h][j]);
  #pragma unroll
  for (int o = 16; o > 0; o >>= 1) m = fmaxf(m, __shfl_xor(m, o, 32));
  float ssum = 0.f;
  for (int j = ln; j < N; j += 32) { float p = __expf(sc[h][j] - m); sc[h][j] = p; ssum += p; }
  #pragma unroll
  for (int o = 16; o > 0; o >>= 1) ssum += __shfl_xor(ssum, o, 32);
  const float inv = 1.f / ssum;
  for (int j = ln; j < N; j += 32) sc[h][j] *= inv;
  __syncthreads();
  float acc = 0.f;
  const f16* vb = V + (size_t)b * N * HIDN + t;
  const int hh = t >> 5;
  for (int j = 0; j < N; ++j) acc += sc[hh][j] * (float)vb[(size_t)j * HIDN];
  Att[(size_t)row * HIDN + t] = (f16)acc;
}

// ---------- GAT: per-node alpha coefficients + per-layer state init --------
__global__ __launch_bounds__(HIDN)
void gat_alpha_init(const float* __restrict__ xw, const float* __restrict__ asrc,
                    const float* __restrict__ adst, float* __restrict__ alps,
                    float* __restrict__ alpd, unsigned* __restrict__ emaxU,
                    float* __restrict__ denom, float* __restrict__ agg) {
  const int row = blockIdx.x, t = threadIdx.x;
  const float x = xw[(size_t)row * HIDN + t];
  const int h = t >> 5, c = t & 31;
  float ps = x * asrc[h * HC + c];
  float pd = x * adst[h * HC + c];
  #pragma unroll
  for (int o = 16; o > 0; o >>= 1) { ps += __shfl_down(ps, o, 32); pd += __shfl_down(pd, o, 32); }
  if (c == 0) {
    alps[row * HEADS + h] = ps;
    alpd[row * HEADS + h] = pd;
    emaxU[row * HEADS + h] = fenc(-1e30f);
    denom[row * HEADS + h] = 0.f;
  }
  agg[(size_t)row * HIDN + t] = 0.f;
}

__device__ __forceinline__ void edge_sd(const int* __restrict__ ei, int b, int e,
                                        int E, int& s, int& d) {
  if (e < E) { s = ei[((size_t)b * 2) * E + e]; d = ei[((size_t)b * 2 + 1) * E + e]; }
  else       { s = d = e - E; }   // self loop
}

__global__ __launch_bounds__(256)
void gat_edge_max(const int* __restrict__ ei, const float* __restrict__ alps,
                  const float* __restrict__ alpd, float* __restrict__ ebuf,
                  unsigned* __restrict__ emaxU, int B, int N, int E, int ETOT) {
  const int idx = blockIdx.x * 256 + threadIdx.x;
  if (idx >= B * ETOT) return;
  const int b = idx / ETOT, e = idx - b * ETOT;
  int s, d; edge_sd(ei, b, e, E, s, d);
  const int rs = b * N + s, rd = b * N + d;
  #pragma unroll
  for (int h = 0; h < HEADS; ++h) {
    float v = alps[rs * HEADS + h] + alpd[rd * HEADS + h];
    v = v > 0.f ? v : 0.2f * v;                  // leaky relu
    ebuf[(size_t)idx * HEADS + h] = v;
    atomicMax(&emaxU[rd * HEADS + h], fenc(v));
  }
}

__global__ __launch_bounds__(256)
void gat_edge_exp(const int* __restrict__ ei, const unsigned* __restrict__ emaxU,
                  float* __restrict__ ebuf, float* __restrict__ denom,
                  int B, int N, int E, int ETOT) {
  const int idx = blockIdx.x * 256 + threadIdx.x;
  if (idx >= B * ETOT) return;
  const int b = idx / ETOT, e = idx - b * ETOT;
  int s, d; edge_sd(ei, b, e, E, s, d);
  const int rd = b * N + d;
  #pragma unroll
  for (int h = 0; h < HEADS; ++h) {
    const float p = __expf(ebuf[(size_t)idx * HEADS + h] - fdec(emaxU[rd * HEADS + h]));
    ebuf[(size_t)idx * HEADS + h] = p;
    atomicAdd(&denom[rd * HEADS + h], p);
  }
}

__global__ __launch_bounds__(HIDN)
void gat_edge_agg(const int* __restrict__ ei, const float* __restrict__ ebuf,
                  const float* __restrict__ denom, const float* __restrict__ xw,
                  float* __restrict__ agg, int B, int N, int E, int ETOT) {
  const int idx = blockIdx.x;
  const int b = idx / ETOT, e = idx - b * ETOT;
  int s, d; edge_sd(ei, b, e, E, s, d);
  const int t = threadIdx.x, h = t >> 5;
  const float alpha = ebuf[(size_t)idx * HEADS + h]
                    / (denom[(b * N + d) * HEADS + h] + 1e-16f);
  atomicAdd(&agg[((size_t)(b * N + d)) * HIDN + t],
            alpha * xw[((size_t)(b * N + s)) * HIDN + t]);
}

__global__ __launch_bounds__(HIDN)
void gat_finalize(const float* __restrict__ agg, const float* __restrict__ gb,
                  f16* __restrict__ Hh, float* __restrict__ Out) {
  const int row = blockIdx.x, t = threadIdx.x;
  const float v = fmaxf(agg[(size_t)row * HIDN + t] + gb[t], 0.f);
  Hh[(size_t)row * HIDN + t] = (f16)v;
  if (Out) Out[(size_t)row * HIDN + t] = v;
}

// ===========================================================================
extern "C" void kernel_launch(void* const* d_in, const int* in_sizes, int n_in,
                              void* d_out, int out_size, void* d_ws, size_t ws_size,
                              hipStream_t stream) {
  (void)n_in; (void)out_size; (void)ws_size;
  const float* dept  = (const float*)d_in[0];
  const float* flow  = (const float*)d_in[1];
  const int*   ei    = (const int*)  d_in[2];
  /* d_in[3] = edge_weight: unused by the reference */
  const int*   mask  = (const int*)  d_in[4];
  const float* dw    = (const float*)d_in[5];
  const float* db    = (const float*)d_in[6];
  const float* dg    = (const float*)d_in[7];
  const float* dbb   = (const float*)d_in[8];
  const float* fw    = (const float*)d_in[9];
  const float* fb    = (const float*)d_in[10];
  const float* fg    = (const float*)d_in[11];
  const float* fbb   = (const float*)d_in[12];
  const float* wq    = (const float*)d_in[13];
  const float* bq    = (const float*)d_in[14];
  const float* wk    = (const float*)d_in[15];
  const float* bk    = (const float*)d_in[16];
  const float* wv    = (const float*)d_in[17];
  const float* bv    = (const float*)d_in[18];
  const float* wo    = (const float*)d_in[19];
  const float* bo    = (const float*)d_in[20];
  const float* fusw  = (const float*)d_in[21];
  const float* fusb  = (const float*)d_in[22];
  const float* fusg  = (const float*)d_in[23];
  const float* fusbb = (const float*)d_in[24];
  const float* gW    = (const float*)d_in[25];
  const float* gas   = (const float*)d_in[26];
  const float* gad   = (const float*)d_in[27];
  const float* gb    = (const float*)d_in[28];

  const int ATTR = 7;
  const int BN = in_sizes[0] / ATTR;       // 51200
  const int N  = in_sizes[1] / BN;         // 100
  const int B  = BN / N;                   // 512
  const int E  = in_sizes[2] / (2 * B);    // 1024
  const int ETOT = E + N;                  // edges + self loops

  // -------- workspace carve-out (256 B aligned) --------
  size_t off = 0;
  auto alloc = [&](size_t bytes) -> void* {
    size_t o = (off + 255) & ~(size_t)255;
    off = o + bytes;
    return (void*)((char*)d_ws + o);
  };
  f16*   cat   = (f16*)alloc((size_t)BN * 256 * sizeof(f16)); // [d | att@wo+bo]
  f16*   fe    = (f16*)alloc((size_t)BN * HIDN * sizeof(f16));
  f16*   flowp = (f16*)alloc((size_t)BN * HIDN * sizeof(f16));
  f16*   qh    = (f16*)alloc((size_t)BN * HIDN * sizeof(f16));
  f16*   kh    = (f16*)alloc((size_t)BN * HIDN * sizeof(f16));
  f16*   vh    = (f16*)alloc((size_t)BN * HIDN * sizeof(f16));
  f16*   atth  = (f16*)alloc((size_t)BN * HIDN * sizeof(f16));
  f16*   hh    = (f16*)alloc((size_t)BN * HIDN * sizeof(f16));
  float* tmp1  = (float*)alloc((size_t)BN * HIDN * sizeof(float)); // gemm f32 / agg
  float* tmp2  = (float*)alloc((size_t)BN * HIDN * sizeof(float)); // xw
  float* alps  = (float*)alloc((size_t)BN * HEADS * sizeof(float));
  float* alpd  = (float*)alloc((size_t)BN * HEADS * sizeof(float));
  unsigned* emaxU = (unsigned*)alloc((size_t)BN * HEADS * sizeof(unsigned));
  float* denom = (float*)alloc((size_t)BN * HEADS * sizeof(float));
  float* ebuf  = (float*)alloc((size_t)B * ETOT * HEADS * sizeof(float));
  f16*   fwt   = (f16*)alloc((size_t)HIDN * HIDN * sizeof(f16));
  f16*   wqt   = (f16*)alloc((size_t)HIDN * HIDN * sizeof(f16));
  f16*   wkt   = (f16*)alloc((size_t)HIDN * HIDN * sizeof(f16));
  f16*   wvt   = (f16*)alloc((size_t)HIDN * HIDN * sizeof(f16));
  f16*   wot   = (f16*)alloc((size_t)HIDN * HIDN * sizeof(f16));
  f16*   fuswt = (f16*)alloc((size_t)HIDN * 256 * sizeof(f16));
  f16*   gwt   = (f16*)alloc((size_t)3 * HIDN * HIDN * sizeof(f16));

  // -------- weight prep (f32 -> f16, transposed to Wt[out][in], K padded) --
  prep_wt<<<HIDN, HIDN, 0, stream>>>(fw,   N,    HIDN, fwt);
  prep_wt<<<HIDN, HIDN, 0, stream>>>(wq,   HIDN, HIDN, wqt);
  prep_wt<<<HIDN, HIDN, 0, stream>>>(wk,   HIDN, HIDN, wkt);
  prep_wt<<<HIDN, HIDN, 0, stream>>>(wv,   HIDN, HIDN, wvt);
  prep_wt<<<HIDN, HIDN, 0, stream>>>(wo,   HIDN, HIDN, wot);
  prep_wt<<<HIDN, HIDN, 0, stream>>>(fusw, 256,  256,  fuswt);
  for (int l = 0; l < 3; ++l)
    prep_wt<<<HIDN, HIDN, 0, stream>>>(gW + (size_t)l * HIDN * HIDN, HIDN, HIDN,
                                       gwt + (size_t)l * HIDN * HIDN);

  // -------- flow encoder --------
  pad_flow_f16<<<BN, HIDN, 0, stream>>>(flow, flowp, N);
  dept_encode<<<BN, HIDN, 0, stream>>>(dept, dw, db, dg, dbb, cat, 256, ATTR);

  const int gblk = (BN + 31) / 32;
  // f = LN(relu(flow @ fw + fb))
  wmma_gemm128<<<gblk, 256, 0, stream>>>(flowp, HIDN, fwt, HIDN, fb, tmp1, nullptr, HIDN, HIDN, BN);
  relu_ln128<<<BN, HIDN, 0, stream>>>(tmp1, fg, fbb, fe, HIDN);
  // q from d (first half of cat), k/v from f
  wmma_gemm128<<<gblk, 256, 0, stream>>>(cat, 256,  wqt, HIDN, bq, nullptr, qh, HIDN, HIDN, BN);
  wmma_gemm128<<<gblk, 256, 0, stream>>>(fe,  HIDN, wkt, HIDN, bk, nullptr, kh, HIDN, HIDN, BN);
  wmma_gemm128<<<gblk, 256, 0, stream>>>(fe,  HIDN, wvt, HIDN, bv, nullptr, vh, HIDN, HIDN, BN);
  // attention
  attn128<<<BN, HIDN, 0, stream>>>(qh, kh, vh, mask, atth, N);
  // att @ wo + bo -> second half of cat (ldc = 256)
  wmma_gemm128<<<gblk, 256, 0, stream>>>(atth, HIDN, wot, HIDN, bo, nullptr, cat + HIDN, 256, HIDN, BN);
  // fused = LN(relu(cat @ fusw + fusb))
  wmma_gemm128<<<gblk, 256, 0, stream>>>(cat, 256, fuswt, 256, fusb, tmp1, nullptr, HIDN, 256, BN);
  relu_ln128<<<BN, HIDN, 0, stream>>>(tmp1, fusg, fusbb, hh, HIDN);

  // -------- 3 GAT layers --------
  const int etotal = B * ETOT;
  for (int l = 0; l < 3; ++l) {
    wmma_gemm128<<<gblk, 256, 0, stream>>>(hh, HIDN, gwt + (size_t)l * HIDN * HIDN,
                                           HIDN, nullptr, tmp2, nullptr, HIDN, HIDN, BN);
    gat_alpha_init<<<BN, HIDN, 0, stream>>>(tmp2, gas + l * HEADS * HC, gad + l * HEADS * HC,
                                            alps, alpd, emaxU, denom, tmp1);
    gat_edge_max<<<(etotal + 255) / 256, 256, 0, stream>>>(ei, alps, alpd, ebuf, emaxU, B, N, E, ETOT);
    gat_edge_exp<<<(etotal + 255) / 256, 256, 0, stream>>>(ei, emaxU, ebuf, denom, B, N, E, ETOT);
    gat_edge_agg<<<etotal, HIDN, 0, stream>>>(ei, ebuf, denom, tmp2, tmp1, B, N, E, ETOT);
    gat_finalize<<<BN, HIDN, 0, stream>>>(tmp1, gb + l * HIDN, hh,
                                          (l == 2) ? (float*)d_out : nullptr);
  }
}
